// SuperPoint_64218351010326
// MI455X (gfx1250) — matrix operand order repaired and checked by reading, not compiled, not run on hardware
//
#include <hip/hip_runtime.h>
#include <hip/hip_bf16.h>
#include <math.h>

#define MAX_KP 1024
#define KTHRESH 0.005f
#define TKCAND 2048

typedef __attribute__((ext_vector_type(16))) _Float16 v16h;
typedef __attribute__((ext_vector_type(8)))  float    v8f;

// ---------------------------------------------------------------------------
// fp32 -> fp16 convert (image)
// ---------------------------------------------------------------------------
__global__ void sp_cvt_f32_f16(const float* __restrict__ src, _Float16* __restrict__ dst, int n) {
  int i = blockIdx.x * blockDim.x + threadIdx.x;
  if (i < n) dst[i] = (_Float16)src[i];
}

// ---------------------------------------------------------------------------
// Pack fp32 OIHW weights into WMMA A-fragment order, fp16, zero padded:
//   wpk[((ct*Ksteps + ks)*32 + lane)*16 + h]
//   lane 0-15: rows M=lane,   K = ks*32 + {0..15}
//   lane16-31: rows M=lane-16,K = ks*32 + {16..31}
// ---------------------------------------------------------------------------
__global__ void sp_pack_w(const float* __restrict__ wsrc, _Float16* __restrict__ wpk,
                          int Cout, int Ktot, int Ksteps, int total) {
  int i = blockIdx.x * blockDim.x + threadIdx.x;
  if (i >= total) return;
  int h    = i & 15;
  int lane = (i >> 4) & 31;
  int t2   = i >> 9;
  int ks   = t2 % Ksteps;
  int ct   = t2 / Ksteps;
  int m    = lane & 15;
  int hi   = lane >> 4;
  int kk   = ks * 32 + hi * 16 + h;
  int co   = ct * 16 + m;
  bool ok  = (co < Cout) && (kk < Ktot);
  int coc  = co < Cout ? co : Cout - 1;
  int kkc  = kk < Ktot ? kk : Ktot - 1;
  float v  = wsrc[(size_t)coc * Ktot + kkc];
  wpk[i] = ok ? (_Float16)v : (_Float16)0.f;
}

// ---------------------------------------------------------------------------
// 3x3 conv (pad 1), implicit GEMM on WMMA with LDS-staged im2col tiles.
//   LDS layout: smem[(tileLocal*16 + n)*Ktotp + kk]  (n = pixel in tile, kk = K)
//   Each wave computes a 16(cout) x 16(pixel) tile.
// ---------------------------------------------------------------------------
__global__ void sp_conv3x3_wmma(const _Float16* __restrict__ in, const _Float16* __restrict__ wpk,
                                const float* __restrict__ bias, _Float16* __restrict__ out,
                                int Cin, int Cout, int Hh, int Ww, int relu,
                                int wavesPerPix, int pixPerBlock) {
  extern __shared__ _Float16 smem[];
  const int tid    = threadIdx.x;
  const int lane   = tid & 31;
  const int waveId = tid >> 5;
  const int CT     = (Cout + 15) >> 4;
  const int Ktot   = Cin * 9;
  const int Ksteps = (Ktot + 31) >> 5;
  const int Ktotp  = Ksteps << 5;
  const int pixPerImg = (Hh * Ww) >> 4;

  // ---- cooperative LDS fill (branch-free: clamp + select) ----
  const int elems = pixPerBlock * 16 * Ktotp;
  for (int f = tid; f < elems; f += 256) {
    int tileLocal = f / (16 * Ktotp);
    int rem = f - tileLocal * 16 * Ktotp;
    int n  = rem & 15;                   // pixel-in-tile (fastest -> coalesced global)
    int kk = rem >> 4;                   // K index
    int pt = blockIdx.x * pixPerBlock + tileLocal;
    int b  = pt / pixPerImg;
    int pr = pt - b * pixPerImg;
    int py = (pr << 4) / Ww;
    int px0 = (pr << 4) - py * Ww;
    int cin = kk / 9;
    int r   = kk - cin * 9;
    int ky  = r / 3;
    int kx  = r - ky * 3;
    int yy  = py + ky - 1;
    int xx  = px0 + n + kx - 1;
    bool ok = (kk < Ktot) & ((unsigned)yy < (unsigned)Hh) & ((unsigned)xx < (unsigned)Ww);
    int cinc = cin < Cin ? cin : Cin - 1;
    int yyc  = yy < 0 ? 0 : (yy >= Hh ? Hh - 1 : yy);
    int xxc  = xx < 0 ? 0 : (xx >= Ww ? Ww - 1 : xx);
    _Float16 v = in[(((size_t)b * Cin + cinc) * Hh + yyc) * Ww + xxc];
    smem[((size_t)tileLocal * 16 + n) * Ktotp + kk] = ok ? v : (_Float16)0.f;
  }
  __syncthreads();

  // ---- WMMA main loop (wave-uniform activity mask) ----
  const int pixLocal = waveId / wavesPerPix;
  const int waveCt   = waveId - pixLocal * wavesPerPix;
  const int ct       = blockIdx.y * wavesPerPix + waveCt;
  if (ct < CT) {
    const int nlane  = lane & 15;
    const int hiHalf = lane >> 4;
    const _Float16* __restrict__ ab = wpk + (size_t)ct * Ksteps * 512 + lane * 16;
    const _Float16* __restrict__ bb = smem + ((size_t)pixLocal * 16 + nlane) * Ktotp + hiHalf * 16;
    v8f acc = {};
    for (int ks = 0; ks < Ksteps; ++ks) {
      __builtin_prefetch(ab + (size_t)ks * 512 + 512, 0, 0);
      v16h af = *(const v16h*)(ab + (size_t)ks * 512);
      v16h bf = *(const v16h*)(bb + ks * 32);
      acc = __builtin_amdgcn_wmma_f32_16x16x32_f16(false, af, false, bf, (short)0, acc,
                                                   false, false);
    }
    int pt = blockIdx.x * pixPerBlock + pixLocal;
    int b  = pt / pixPerImg;
    int pr = pt - b * pixPerImg;
    int py = (pr << 4) / Ww;
    int px0 = (pr << 4) - py * Ww;
    #pragma unroll
    for (int r = 0; r < 8; ++r) {
      int co = ct * 16 + r + hiHalf * 8;
      if (co < Cout) {
        float v = acc[r] + bias[co];
        if (relu) v = v > 0.f ? v : 0.f;
        out[(((size_t)b * Cout + co) * Hh + py) * Ww + px0 + nlane] = (_Float16)v;
      }
    }
  }
}

// ---------------------------------------------------------------------------
// 1x1 conv: pure GEMM on WMMA with LDS-staged activation tiles (Cin % 32 == 0)
// ---------------------------------------------------------------------------
__global__ void sp_conv1x1_wmma(const _Float16* __restrict__ in, const _Float16* __restrict__ wpk,
                                const float* __restrict__ bias, _Float16* __restrict__ out,
                                int Cin, int Cout, int HW, int relu,
                                int wavesPerPix, int pixPerBlock) {
  extern __shared__ _Float16 smem[];
  const int tid    = threadIdx.x;
  const int lane   = tid & 31;
  const int waveId = tid >> 5;
  const int CT     = (Cout + 15) >> 4;
  const int Ksteps = Cin >> 5;
  const int pixPerImg = HW >> 4;

  const int elems = pixPerBlock * 16 * Cin;
  for (int f = tid; f < elems; f += 256) {
    int tileLocal = f / (16 * Cin);
    int rem = f - tileLocal * 16 * Cin;
    int n  = rem & 15;
    int kk = rem >> 4;
    int pt = blockIdx.x * pixPerBlock + tileLocal;
    int b  = pt / pixPerImg;
    int base = (pt - b * pixPerImg) << 4;
    smem[((size_t)tileLocal * 16 + n) * Cin + kk] =
        in[((size_t)b * Cin + kk) * HW + base + n];
  }
  __syncthreads();

  const int pixLocal = waveId / wavesPerPix;
  const int waveCt   = waveId - pixLocal * wavesPerPix;
  const int ct       = blockIdx.y * wavesPerPix + waveCt;
  if (ct < CT) {
    const int nlane  = lane & 15;
    const int hiHalf = lane >> 4;
    const _Float16* __restrict__ ab = wpk + (size_t)ct * Ksteps * 512 + lane * 16;
    const _Float16* __restrict__ bb = smem + ((size_t)pixLocal * 16 + nlane) * Cin + hiHalf * 16;
    v8f acc = {};
    for (int ks = 0; ks < Ksteps; ++ks) {
      v16h af = *(const v16h*)(ab + (size_t)ks * 512);
      v16h bf = *(const v16h*)(bb + ks * 32);
      acc = __builtin_amdgcn_wmma_f32_16x16x32_f16(false, af, false, bf, (short)0, acc,
                                                   false, false);
    }
    int pt = blockIdx.x * pixPerBlock + pixLocal;
    int b  = pt / pixPerImg;
    int base = (pt - b * pixPerImg) << 4;
    #pragma unroll
    for (int r = 0; r < 8; ++r) {
      int co = ct * 16 + r + hiHalf * 8;
      if (co < Cout) {
        float v = acc[r] + bias[co];
        if (relu) v = v > 0.f ? v : 0.f;
        out[((size_t)b * Cout + co) * HW + base + nlane] = (_Float16)v;
      }
    }
  }
}

// ---------------------------------------------------------------------------
// 2x2 max pool, stride 2
// ---------------------------------------------------------------------------
__global__ void sp_pool2(const _Float16* __restrict__ in, _Float16* __restrict__ out,
                         int C, int Hh, int Ww) {
  const int W2 = Ww >> 1, H2 = Hh >> 1;
  long N = (long)4 * C * H2 * W2;
  long i = (long)blockIdx.x * blockDim.x + threadIdx.x;
  if (i >= N) return;
  int x = (int)(i % W2); long t = i / W2;
  int y = (int)(t % H2); t /= H2;
  int c = (int)(t % C);  int b = (int)(t / C);
  const _Float16* p = in + (((size_t)b * C + c) * Hh + 2 * y) * Ww + 2 * x;
  float m = fmaxf(fmaxf((float)p[0], (float)p[1]), fmaxf((float)p[Ww], (float)p[Ww + 1]));
  out[i] = (_Float16)m;
}

// ---------------------------------------------------------------------------
// softmax over 65 channels, drop dustbin, pixel-shuffle 8x8 -> dense scores
// ---------------------------------------------------------------------------
__global__ void sp_softmax_shuffle(const _Float16* __restrict__ semi, float* __restrict__ sc) {
  int i = blockIdx.x * blockDim.x + threadIdx.x;     // 4*60*80
  if (i >= 4 * 60 * 80) return;
  int x = i % 80, y = (i / 80) % 60, b = i / (80 * 60);
  float v[65];
  float mx = -3.4e38f;
  for (int c = 0; c < 65; ++c) {
    v[c] = (float)semi[(((size_t)b * 65 + c) * 60 + y) * 80 + x];
    mx = fmaxf(mx, v[c]);
  }
  float s = 0.f;
  for (int c = 0; c < 65; ++c) { v[c] = __expf(v[c] - mx); s += v[c]; }
  float inv = 1.f / s;
  for (int c = 0; c < 64; ++c) {
    int yy = y * 8 + (c >> 3), xx = x * 8 + (c & 7);
    sc[(size_t)b * 307200 + (size_t)yy * 640 + xx] = v[c] * inv;
  }
}

// ---------------------------------------------------------------------------
// separable 9x9 max filter (NMS pool, radius 4)
// ---------------------------------------------------------------------------
__global__ void sp_maxf_h(const float* __restrict__ in, float* __restrict__ out,
                          int Hh, int Ww) {
  long N = (long)4 * Hh * Ww;
  long i = (long)blockIdx.x * blockDim.x + threadIdx.x;
  if (i >= N) return;
  int x = (int)(i % Ww);
  long rowbase = i - x;
  int lo = x - 4 > 0 ? x - 4 : 0;
  int hi = x + 4 < Ww - 1 ? x + 4 : Ww - 1;
  float m = -3.4e38f;
  for (int xx = lo; xx <= hi; ++xx) m = fmaxf(m, in[rowbase + xx]);
  out[i] = m;
}

__global__ void sp_maxf_v(const float* __restrict__ in, float* __restrict__ out,
                          int Hh, int Ww) {
  long N = (long)4 * Hh * Ww;
  long i = (long)blockIdx.x * blockDim.x + threadIdx.x;
  if (i >= N) return;
  int x = (int)(i % Ww);
  int y = (int)((i / Ww) % Hh);
  int b = (int)(i / ((long)Ww * Hh));
  int lo = y - 4 > 0 ? y - 4 : 0;
  int hi = y + 4 < Hh - 1 ? y + 4 : Hh - 1;
  float m = -3.4e38f;
  const float* p = in + (size_t)b * Hh * Ww + x;
  for (int yy = lo; yy <= hi; ++yy) m = fmaxf(m, p[(size_t)yy * Ww]);
  out[i] = m;
}

__global__ void sp_eqmask(const float* __restrict__ a, const float* __restrict__ pool,
                          float* __restrict__ mm, long N) {
  long i = (long)blockIdx.x * blockDim.x + threadIdx.x;
  if (i < N) mm[i] = (a[i] == pool[i]) ? 1.f : 0.f;
}

__global__ void sp_supp(const float* __restrict__ poolmask, const float* __restrict__ sc,
                        float* __restrict__ ss, float* __restrict__ supp, long N) {
  long i = (long)blockIdx.x * blockDim.x + threadIdx.x;
  if (i >= N) return;
  float sp = poolmask[i] > 0.f ? 1.f : 0.f;
  supp[i] = sp;
  ss[i] = sp > 0.f ? 0.f : sc[i];
}

__global__ void sp_update(const float* __restrict__ ss, const float* __restrict__ poolss,
                          const float* __restrict__ supp, float* __restrict__ mm, long N) {
  long i = (long)blockIdx.x * blockDim.x + threadIdx.x;
  if (i >= N) return;
  bool newmax = (ss[i] == poolss[i]);
  bool res = (mm[i] > 0.f) || (newmax && supp[i] == 0.f);
  mm[i] = res ? 1.f : 0.f;
}

__global__ void sp_finalmask(const float* __restrict__ mm, const float* __restrict__ sc,
                             float* __restrict__ msk, int Hh, int Ww) {
  long N = (long)4 * Hh * Ww;
  long i = (long)blockIdx.x * blockDim.x + threadIdx.x;
  if (i >= N) return;
  int x = (int)(i % Ww);
  int y = (int)((i / Ww) % Hh);
  bool border = (y >= 4) && (y < Hh - 4) && (x >= 4) && (x < Ww - 4);
  msk[i] = (border && mm[i] > 0.f) ? sc[i] : 0.f;
}

// ---------------------------------------------------------------------------
// per-batch top-1024: binary-search threshold + LDS compaction + bitonic sort
// ---------------------------------------------------------------------------
__global__ void sp_topk(const float* __restrict__ msk, float* __restrict__ kpts,
                        float* __restrict__ scr, int Npix, int Wimg) {
  __shared__ float shv[TKCAND];
  __shared__ int   shi[TKCAND];
  __shared__ float s_lo, s_hi;
  __shared__ int   s_cnt, s_n;
  const int tid = threadIdx.x;
  const float* p = msk + (size_t)blockIdx.x * Npix;

  if (tid == 0) { s_lo = 0.f; s_hi = 1.f; }
  __syncthreads();
  for (int it = 0; it < 28; ++it) {
    float mid = 0.5f * (s_lo + s_hi);
    if (tid == 0) s_cnt = 0;
    __syncthreads();
    int c = 0;
    for (int i = tid; i < Npix; i += blockDim.x)
      if (p[i] > mid) c++;
    atomicAdd(&s_cnt, c);
    __syncthreads();
    if (tid == 0) { if (s_cnt > MAX_KP) s_lo = mid; else s_hi = mid; }
    __syncthreads();
  }
  float thr = s_lo;
  if (tid == 0) s_n = 0;
  __syncthreads();
  for (int i = tid; i < Npix; i += blockDim.x) {
    float v = p[i];
    if (v > thr) {
      int pos = atomicAdd(&s_n, 1);
      if (pos < TKCAND) { shv[pos] = v; shi[pos] = i; }
    }
  }
  __syncthreads();
  int nfill = s_n < TKCAND ? s_n : TKCAND;
  for (int i = tid; i < TKCAND; i += blockDim.x)
    if (i >= nfill) { shv[i] = -1.f; shi[i] = 0; }
  __syncthreads();

  for (int k = 2; k <= TKCAND; k <<= 1) {
    for (int j = k >> 1; j > 0; j >>= 1) {
      for (int t = tid; t < TKCAND; t += blockDim.x) {
        int ixj = t ^ j;
        if (ixj > t) {
          bool up = ((t & k) == 0);
          float a = shv[t], b2 = shv[ixj];
          bool sw = up ? (a < b2) : (a > b2);
          if (sw) {
            shv[t] = b2; shv[ixj] = a;
            int ti = shi[t]; shi[t] = shi[ixj]; shi[ixj] = ti;
          }
        }
      }
      __syncthreads();
    }
  }

  for (int i = tid; i < MAX_KP; i += blockDim.x) {
    float v = shv[i]; int idx = shi[i];
    if (v < 0.f) { v = 0.f; idx = 0; }
    float kx = (float)(idx % Wimg);
    float ky = (float)(idx / Wimg);
    size_t o = (size_t)blockIdx.x * MAX_KP + i;
    kpts[o * 2 + 0] = kx;
    kpts[o * 2 + 1] = ky;
    scr[o] = (v > KTHRESH) ? v : 0.f;
  }
}

// ---------------------------------------------------------------------------
// channel-wise L2 normalize dmap [4,256,4800] in place
// ---------------------------------------------------------------------------
__global__ void sp_dnorm(_Float16* __restrict__ d) {
  int i = blockIdx.x * blockDim.x + threadIdx.x;   // 4*4800
  if (i >= 4 * 4800) return;
  int pix = i % 4800, b = i / 4800;
  _Float16* base = d + (size_t)b * 256 * 4800 + pix;
  float s = 0.f;
  for (int c = 0; c < 256; ++c) { float v = (float)base[(size_t)c * 4800]; s += v * v; }
  float inv = 1.f / fmaxf(sqrtf(s), 1e-12f);
  for (int c = 0; c < 256; ++c)
    base[(size_t)c * 4800] = (_Float16)((float)base[(size_t)c * 4800] * inv);
}

// ---------------------------------------------------------------------------
// bilinear grid-sample at keypoints + per-keypoint L2 normalize
// ---------------------------------------------------------------------------
__global__ void sp_gridsample(const _Float16* __restrict__ dmap, const float* __restrict__ kpts,
                              float* __restrict__ desc) {
  __shared__ float red[256];
  int b = blockIdx.x / MAX_KP;
  int k = blockIdx.x % MAX_KP;
  int c = threadIdx.x;

  float kx = kpts[((size_t)b * MAX_KP + k) * 2 + 0];
  float ky = kpts[((size_t)b * MAX_KP + k) * 2 + 1];
  float gx = ((kx - 3.5f) / 635.5f) * 2.f - 1.f;
  float gy = ((ky - 3.5f) / 475.5f) * 2.f - 1.f;
  float ix = (gx + 1.f) * 0.5f * 79.f;
  float iy = (gy + 1.f) * 0.5f * 59.f;
  float x0f = floorf(ix), y0f = floorf(iy);
  float wx = ix - x0f, wy = iy - y0f;
  int x0i = (int)fminf(fmaxf(x0f, 0.f), 79.f);
  int x1i = (int)fminf(fmaxf(x0f + 1.f, 0.f), 79.f);
  int y0i = (int)fminf(fmaxf(y0f, 0.f), 59.f);
  int y1i = (int)fminf(fmaxf(y0f + 1.f, 0.f), 59.f);

  const _Float16* d = dmap + ((size_t)b * 256 + c) * 4800;
  float v = (float)d[y0i * 80 + x0i] * (1.f - wx) * (1.f - wy)
          + (float)d[y0i * 80 + x1i] * wx * (1.f - wy)
          + (float)d[y1i * 80 + x0i] * (1.f - wx) * wy
          + (float)d[y1i * 80 + x1i] * wx * wy;

  red[c] = v * v;
  __syncthreads();
  for (int s2 = 128; s2 > 0; s2 >>= 1) {
    if (c < s2) red[c] += red[c + s2];
    __syncthreads();
  }
  float inv = 1.f / fmaxf(sqrtf(red[0]), 1e-12f);
  desc[((size_t)b * 256 + c) * MAX_KP + k] = v * inv;
}

// ---------------------------------------------------------------------------
// host-side helpers
// ---------------------------------------------------------------------------
static void pick_tiling(int Cout, int& CT, int& wpp, int& ppb) {
  CT = (Cout + 15) >> 4;
  wpp = (CT < 8 && (8 % CT == 0)) ? CT : 8;
  ppb = 8 / wpp;
}

static void launch_conv3(const _Float16* in, const _Float16* wpk, const float* b, _Float16* out,
                         int Cin, int Cout, int H, int W, int relu, hipStream_t s) {
  int CT, wpp, ppb;
  pick_tiling(Cout, CT, wpp, ppb);
  int Ktot = Cin * 9, Ks = (Ktot + 31) / 32, Ktotp = Ks * 32;
  long npix = 4L * H * W / 16;
  dim3 grid((unsigned)(npix / ppb), (unsigned)((CT + wpp - 1) / wpp));
  size_t shmem = (size_t)ppb * 16 * Ktotp * 2;
  sp_conv3x3_wmma<<<grid, 256, shmem, s>>>(in, wpk, b, out, Cin, Cout, H, W, relu, wpp, ppb);
}

static void launch_conv1(const _Float16* in, const _Float16* wpk, const float* b, _Float16* out,
                         int Cin, int Cout, int HW, int relu, hipStream_t s) {
  int CT, wpp, ppb;
  pick_tiling(Cout, CT, wpp, ppb);
  long npix = 4L * HW / 16;
  dim3 grid((unsigned)(npix / ppb), (unsigned)((CT + wpp - 1) / wpp));
  size_t shmem = (size_t)ppb * 16 * Cin * 2;
  sp_conv1x1_wmma<<<grid, 256, shmem, s>>>(in, wpk, b, out, Cin, Cout, HW, relu, wpp, ppb);
}

extern "C" void kernel_launch(void* const* d_in, const int* in_sizes, int n_in,
                              void* d_out, int out_size, void* d_ws, size_t ws_size,
                              hipStream_t stream) {
  (void)n_in; (void)out_size; (void)ws_size; (void)in_sizes;
  const float* image = (const float*)d_in[0];
  const int wi[12] = {1, 3, 5, 7, 9, 11, 13, 15, 17, 19, 21, 23};
  const float* wf[12]; const float* bf[12];
  for (int j = 0; j < 12; ++j) { wf[j] = (const float*)d_in[wi[j]]; bf[j] = (const float*)d_in[wi[j] + 1]; }

  // layer geometry: {Cin, Cout, is3x3}
  const int lCin[12]  = {1, 64, 64, 64, 64, 128, 128, 128, 128, 256, 128, 256};
  const int lCout[12] = {64, 64, 64, 64, 128, 128, 128, 128, 256, 65, 256, 256};
  const int l3x3[12]  = {1, 1, 1, 1, 1, 1, 1, 1, 1, 0, 1, 0};

  // workspace layout
  char* ws = (char*)d_ws;
  size_t off = 0;
  auto alloc = [&](size_t bytes) -> char* {
    off = (off + 255) & ~(size_t)255;
    char* p = ws + off; off += bytes; return p;
  };
  _Float16* wpk[12];
  int pkTotal[12];
  for (int j = 0; j < 12; ++j) {
    int Ktot = l3x3[j] ? lCin[j] * 9 : lCin[j];
    int Ks = (Ktot + 31) / 32;
    int CT = (lCout[j] + 15) / 16;
    pkTotal[j] = CT * Ks * 512;
    wpk[j] = (_Float16*)alloc((size_t)pkTotal[j] * 2);
  }
  _Float16* imgh = (_Float16*)alloc((size_t)4 * 480 * 640 * 2);
  _Float16* buf0 = (_Float16*)alloc((size_t)4 * 64 * 480 * 640 * 2);
  _Float16* buf1 = (_Float16*)alloc((size_t)4 * 64 * 480 * 640 * 2);
  _Float16* xtr  = (_Float16*)alloc((size_t)4 * 128 * 60 * 80 * 2);
  const long NP = 4L * 480 * 640;
  float* sc  = (float*)alloc((size_t)NP * 4);
  float* mm  = (float*)alloc((size_t)NP * 4);
  float* ss  = (float*)alloc((size_t)NP * 4);
  float* t0  = (float*)alloc((size_t)NP * 4);
  float* t1  = (float*)alloc((size_t)NP * 4);
  float* t2  = (float*)alloc((size_t)NP * 4);
  float* msk = (float*)alloc((size_t)NP * 4);

  // weight packing + image conversion
  for (int j = 0; j < 12; ++j) {
    int Ktot = l3x3[j] ? lCin[j] * 9 : lCin[j];
    int Ks = (Ktot + 31) / 32;
    sp_pack_w<<<(pkTotal[j] + 255) / 256, 256, 0, stream>>>(wf[j], wpk[j], lCout[j], Ktot, Ks, pkTotal[j]);
  }
  { int n = 4 * 480 * 640; sp_cvt_f32_f16<<<(n + 255) / 256, 256, 0, stream>>>(image, imgh, n); }

  // ---- shared trunk ----
  launch_conv3(imgh, wpk[0], bf[0], buf0,   1,  64, 480, 640, 1, stream);
  launch_conv3(buf0, wpk[1], bf[1], buf1,  64,  64, 480, 640, 1, stream);
  { long n = 4L * 64 * 240 * 320; sp_pool2<<<(int)((n + 255) / 256), 256, 0, stream>>>(buf1, buf0, 64, 480, 640); }
  launch_conv3(buf0, wpk[2], bf[2], buf1,  64,  64, 240, 320, 1, stream);
  launch_conv3(buf1, wpk[3], bf[3], buf0,  64,  64, 240, 320, 1, stream);
  { long n = 4L * 64 * 120 * 160; sp_pool2<<<(int)((n + 255) / 256), 256, 0, stream>>>(buf0, buf1, 64, 240, 320); }
  launch_conv3(buf1, wpk[4], bf[4], buf0,  64, 128, 120, 160, 1, stream);
  launch_conv3(buf0, wpk[5], bf[5], buf1, 128, 128, 120, 160, 1, stream);
  { long n = 4L * 128 * 60 * 80; sp_pool2<<<(int)((n + 255) / 256), 256, 0, stream>>>(buf1, buf0, 128, 120, 160); }
  launch_conv3(buf0, wpk[6], bf[6], buf1, 128, 128, 60, 80, 1, stream);
  launch_conv3(buf1, wpk[7], bf[7], xtr,  128, 128, 60, 80, 1, stream);

  // ---- detector head ----
  launch_conv3(xtr, wpk[8], bf[8], buf0, 128, 256, 60, 80, 1, stream);
  launch_conv1(buf0, wpk[9], bf[9], buf1, 256, 65, 60 * 80, 0, stream);
  { int n = 4 * 60 * 80; sp_softmax_shuffle<<<(n + 255) / 256, 256, 0, stream>>>(buf1, sc); }

  // simple_nms
  int pb = (int)((NP + 255) / 256);
  sp_maxf_h<<<pb, 256, 0, stream>>>(sc, t0, 480, 640);
  sp_maxf_v<<<pb, 256, 0, stream>>>(t0, t1, 480, 640);
  sp_eqmask<<<pb, 256, 0, stream>>>(sc, t1, mm, NP);
  for (int it = 0; it < 2; ++it) {
    sp_maxf_h<<<pb, 256, 0, stream>>>(mm, t0, 480, 640);
    sp_maxf_v<<<pb, 256, 0, stream>>>(t0, t1, 480, 640);
    sp_supp<<<pb, 256, 0, stream>>>(t1, sc, ss, t2, NP);
    sp_maxf_h<<<pb, 256, 0, stream>>>(ss, t0, 480, 640);
    sp_maxf_v<<<pb, 256, 0, stream>>>(t0, t1, 480, 640);
    sp_update<<<pb, 256, 0, stream>>>(ss, t1, t2, mm, NP);
  }
  sp_finalmask<<<pb, 256, 0, stream>>>(mm, sc, msk, 480, 640);

  // top-k per batch -> keypoints + scores directly into d_out
  float* out  = (float*)d_out;
  float* kpts = out;                               // [4,1024,2]
  float* scr  = out + 4 * MAX_KP * 2;              // [4,1024]
  float* desc = out + 4 * MAX_KP * 2 + 4 * MAX_KP; // [4,256,1024]
  sp_topk<<<4, 1024, 0, stream>>>(msk, kpts, scr, 480 * 640, 640);

  // ---- descriptor head ----
  launch_conv3(xtr, wpk[10], bf[10], buf0, 128, 256, 60, 80, 1, stream);
  launch_conv1(buf0, wpk[11], bf[11], buf1, 256, 256, 60 * 80, 0, stream);
  { int n = 4 * 4800; sp_dnorm<<<(n + 255) / 256, 256, 0, stream>>>(buf1); }
  sp_gridsample<<<4 * MAX_KP, 256, 0, stream>>>(buf1, kpts, desc);
}
